// ChannelAttention1D_4054449128044
// MI455X (gfx1250) — matrix operand order repaired and checked by reading, not compiled, not run on hardware
//
#include <hip/hip_runtime.h>
#include <hip/hip_bf16.h>
#include <stdint.h>

// CDNA5 / gfx1250: wave32, WMMA 16x16x32 bf16 -> f32 accumulate
typedef __bf16 v16bf __attribute__((ext_vector_type(16)));
typedef float  v8f   __attribute__((ext_vector_type(8)));

#define B_DIM 8
#define C_DIM 256
#define T_DIM 16384
#define KCHUNKS 16
#define KCHUNK_LEN (T_DIM / KCHUNKS)   // 1024

// float -> bf16 with round-to-nearest-even (bits in low 16 of return)
__device__ __forceinline__ uint32_t f2bf_u(float f) {
    uint32_t u = __float_as_uint(f);
    return (u + 0x7FFFu + ((u >> 16) & 1u)) >> 16;
}
__device__ __forceinline__ uint32_t pack2(float lo, float hi) {
    return f2bf_u(lo) | (f2bf_u(hi) << 16);
}

union FragBF { uint32_t u[8]; v16bf v; };

// ---------------------------------------------------------------------------
// Kernel 0: zero the energy accumulator (K-chunks accumulate via atomics)
// ---------------------------------------------------------------------------
__global__ void zero_f32(float* __restrict__ p, int n) {
    int i = blockIdx.x * 256 + threadIdx.x;
    if (i < n) p[i] = 0.0f;
}

// ---------------------------------------------------------------------------
// Kernel 1: energy[b,i,j] += sum_{k in chunk} x[b,i,k]*x[b,j,k]
// Symmetric: only the 3 lower-triangular 128x128 blocks per batch are
// computed; the off-diagonal block mirrors its atomics into the upper half.
// Block = 128x128 output tile, 8 waves; wave w owns a 2x4 grid of 16x16
// tiles (mi = 2*(w&3)+{0,1}, nj = 4*(w>>2)+{0..3}) -> 8 WMMAs per K-step.
// ---------------------------------------------------------------------------
__global__ __launch_bounds__(256) void gram_kernel(const float* __restrict__ x,
                                                   float* __restrict__ energy) {
    __shared__ uint32_t As[128][17];   // 128 rows x 16 bf16-pairs (+1 pad)
    __shared__ uint32_t Bs[128][17];

    const int tid = threadIdx.x;
    const int bid = blockIdx.x;
    const int b   = bid / (3 * KCHUNKS);
    const int rem = bid % (3 * KCHUNKS);
    const int kc  = rem / 3;
    const int tri = rem % 3;               // 0:(0,0) 1:(1,0) 2:(1,1)
    const int ti  = (tri + 1) >> 1;        // 0,1,1
    const int tj  = tri >> 1;              // 0,0,1
    const int i0  = ti * 128;
    const int j0  = tj * 128;
    const bool offdiag = (ti != tj);       // block-uniform

    const int kBeg = kc * KCHUNK_LEN;
    const int kEnd = kBeg + KCHUNK_LEN;

    const int w    = tid >> 5;
    const int lane = tid & 31;
    const int l15  = lane & 15;
    const int half = lane >> 4;
    const int mi0  = (w & 3) * 2;          // two mi tiles
    const int njb  = (w >> 2) * 4;         // four nj tiles

    v8f acc[2][4];
#pragma unroll
    for (int m = 0; m < 2; ++m)
#pragma unroll
        for (int q = 0; q < 4; ++q) acc[m][q] = (v8f){0,0,0,0,0,0,0,0};

    const int lr = tid >> 1;               // staging row 0..127
    const int lc = (tid & 1) * 16;         // staging col base (16 floats/thread)

    const float* xb = x + (size_t)b * C_DIM * T_DIM;

    for (int k0 = kBeg; k0 < kEnd; k0 += 32) {
        // ---- stage A (rows i0..) and B (rows j0..), fp32 -> bf16 pairs
        {
            const int cp = lc >> 1;
            const float4* pa = reinterpret_cast<const float4*>(
                xb + (size_t)(i0 + lr) * T_DIM + k0 + lc);
            float4 a0 = pa[0], a1 = pa[1], a2 = pa[2], a3 = pa[3];
            As[lr][cp + 0] = pack2(a0.x, a0.y);
            As[lr][cp + 1] = pack2(a0.z, a0.w);
            As[lr][cp + 2] = pack2(a1.x, a1.y);
            As[lr][cp + 3] = pack2(a1.z, a1.w);
            As[lr][cp + 4] = pack2(a2.x, a2.y);
            As[lr][cp + 5] = pack2(a2.z, a2.w);
            As[lr][cp + 6] = pack2(a3.x, a3.y);
            As[lr][cp + 7] = pack2(a3.z, a3.w);
            const float4* pb = reinterpret_cast<const float4*>(
                xb + (size_t)(j0 + lr) * T_DIM + k0 + lc);
            float4 b0 = pb[0], b1 = pb[1], b2 = pb[2], b3 = pb[3];
            Bs[lr][cp + 0] = pack2(b0.x, b0.y);
            Bs[lr][cp + 1] = pack2(b0.z, b0.w);
            Bs[lr][cp + 2] = pack2(b1.x, b1.y);
            Bs[lr][cp + 3] = pack2(b1.z, b1.w);
            Bs[lr][cp + 4] = pack2(b2.x, b2.y);
            Bs[lr][cp + 5] = pack2(b2.z, b2.w);
            Bs[lr][cp + 6] = pack2(b3.x, b3.y);
            Bs[lr][cp + 7] = pack2(b3.z, b3.w);
        }
        __syncthreads();

        // ---- fragments per documented VGPR layouts
        FragBF a[2], bf[4];
        const int kp0 = half * 4;          // A: K pairs by lane half
#pragma unroll
        for (int m = 0; m < 2; ++m) {
            const int ar = (mi0 + m) * 16 + l15;
#pragma unroll
            for (int v = 0; v < 4; ++v) {
                a[m].u[v]     = As[ar][kp0 + v];
                a[m].u[4 + v] = As[ar][8 + kp0 + v];
            }
        }
        const int kp = half * 4;           // B: K pairs by lane half (same as A)
#pragma unroll
        for (int q = 0; q < 4; ++q) {
            const int br = (njb + q) * 16 + l15;
#pragma unroll
            for (int v = 0; v < 8; ++v) bf[q].u[v] = Bs[br][kp + v + (v >= 4 ? 4 : 0)];
        }
#pragma unroll
        for (int m = 0; m < 2; ++m)
#pragma unroll
            for (int q = 0; q < 4; ++q)
                acc[m][q] = __builtin_amdgcn_wmma_f32_16x16x32_bf16(
                    false, a[m].v, false, bf[q].v, (short)0, acc[m][q], false, false);
        __syncthreads();
    }

    // ---- accumulate partials (uniform branch hoisted out of store loops)
    float* Eb = energy + (size_t)b * C_DIM * C_DIM;
#pragma unroll
    for (int m = 0; m < 2; ++m)
#pragma unroll
        for (int q = 0; q < 4; ++q)
#pragma unroll
            for (int r = 0; r < 8; ++r) {
                const int row = i0 + (mi0 + m) * 16 + r + half * 8;
                const int col = j0 + (njb + q) * 16 + l15;
                atomicAdd(&Eb[row * C_DIM + col], acc[m][q][r]);
            }
    if (offdiag) {
        // mirror the whole block across the diagonal
#pragma unroll
        for (int m = 0; m < 2; ++m)
#pragma unroll
            for (int q = 0; q < 4; ++q)
#pragma unroll
                for (int r = 0; r < 8; ++r) {
                    const int row = i0 + (mi0 + m) * 16 + r + half * 8;
                    const int col = j0 + (njb + q) * 16 + l15;
                    atomicAdd(&Eb[col * C_DIM + row], acc[m][q][r]);
                }
    }
}

// ---------------------------------------------------------------------------
// Kernel 2: attention[row, j] = softmax_j(max_j(e) - e) == exp(min_e - e)/sum
// One block per (b, i) row; write bf16.
// ---------------------------------------------------------------------------
__global__ __launch_bounds__(256) void softmax_kernel(const float* __restrict__ energy,
                                                      uint16_t* __restrict__ att) {
    __shared__ float red[256];
    const int tid = threadIdx.x;
    const size_t row = blockIdx.x;      // 0 .. B*C-1
    const float e = energy[row * C_DIM + tid];

    red[tid] = e; __syncthreads();
    for (int s = 128; s > 0; s >>= 1) {
        if (tid < s) red[tid] = fminf(red[tid], red[tid + s]);
        __syncthreads();
    }
    const float emin = red[0];
    __syncthreads();

    const float p = __expf(emin - e);   // <= 1, numerically stable
    red[tid] = p; __syncthreads();
    for (int s = 128; s > 0; s >>= 1) {
        if (tid < s) red[tid] += red[tid + s];
        __syncthreads();
    }
    const float inv = 1.0f / red[0];
    att[row * C_DIM + tid] = (uint16_t)f2bf_u(p * inv);
}

// ---------------------------------------------------------------------------
// Kernel 3: out[b,i,t] = gamma * sum_j att[b,i,j]*x[b,j,t] + x[b,i,t]
// Block = 64(i) x 256(t) tile. X tile staged TRANSPOSED so K(j)-pairs are
// contiguous in LDS for B fragments. Wave w: mi = w&3, nt in [8*(w>>2), +8).
// ---------------------------------------------------------------------------
__global__ __launch_bounds__(256) void apply_kernel(const float* __restrict__ x,
                                                    const uint16_t* __restrict__ att,
                                                    const float* __restrict__ gamma,
                                                    float* __restrict__ out) {
    __shared__ uint32_t As[64][17];     // attention: 64 i-rows x 16 j-pairs
    __shared__ uint32_t Bt[256][17];    // x^T tile : 256 t-rows x 16 j-pairs

    const int tid = threadIdx.x;
    const int bid = blockIdx.x;
    const int b   = bid >> 8;           // 8 batches
    const int rem = bid & 255;
    const int i0  = (rem >> 6) * 64;    // 4 i-tiles
    const int t0  = (rem & 63) * 256;   // 64 t-tiles

    const int w    = tid >> 5;
    const int lane = tid & 31;
    const int l15  = lane & 15;
    const int half = lane >> 4;
    const int mi   = w & 3;
    const int ntb  = (w >> 2) * 8;

    v8f acc[8];
#pragma unroll
    for (int i = 0; i < 8; ++i) acc[i] = (v8f){0,0,0,0,0,0,0,0};

    const float g = gamma[0];
    const float*    xb = x   + (size_t)b * C_DIM * T_DIM;
    const uint16_t* ab = att + (size_t)b * C_DIM * C_DIM;

    const int lr = tid >> 2;
    const int lc = (tid & 3) * 8;

    for (int j0 = 0; j0 < C_DIM; j0 += 32) {
        // ---- stage attention tile (already bf16): 8 ushorts = 1 uint4/thread
        {
            const uint4* pa = reinterpret_cast<const uint4*>(
                ab + (size_t)(i0 + lr) * C_DIM + j0 + lc);
            uint4 av = *pa;
            As[lr][(lc >> 1) + 0] = av.x;
            As[lr][(lc >> 1) + 1] = av.y;
            As[lr][(lc >> 1) + 2] = av.z;
            As[lr][(lc >> 1) + 3] = av.w;
        }
        // ---- stage x tile transposed: Bt[t][j-pair], fp32 -> bf16
#pragma unroll
        for (int kk = 0; kk < 4; ++kk) {
            const int u   = tid + 256 * kk;
            const int jpp = u & 15;        // which j-pair (0..15)
            const int tq  = u >> 4;        // which t-quad (0..63)
            const float4 v0 = *reinterpret_cast<const float4*>(
                xb + (size_t)(j0 + 2 * jpp)     * T_DIM + t0 + 4 * tq);
            const float4 v1 = *reinterpret_cast<const float4*>(
                xb + (size_t)(j0 + 2 * jpp + 1) * T_DIM + t0 + 4 * tq);
            Bt[4 * tq + 0][jpp] = pack2(v0.x, v1.x);
            Bt[4 * tq + 1][jpp] = pack2(v0.y, v1.y);
            Bt[4 * tq + 2][jpp] = pack2(v0.z, v1.z);
            Bt[4 * tq + 3][jpp] = pack2(v0.w, v1.w);
        }
        __syncthreads();

        FragBF a;
        const int ar  = mi * 16 + l15;
        const int kp0 = half * 4;
#pragma unroll
        for (int v = 0; v < 4; ++v) {
            a.u[v]     = As[ar][kp0 + v];
            a.u[4 + v] = As[ar][8 + kp0 + v];
        }
        const int kp = half * 4;
#pragma unroll
        for (int nt = 0; nt < 8; ++nt) {
            FragBF bf;
            const int trow = (ntb + nt) * 16 + l15;
#pragma unroll
            for (int v = 0; v < 8; ++v) bf.u[v] = Bt[trow][kp + v + (v >= 4 ? 4 : 0)];
            acc[nt] = __builtin_amdgcn_wmma_f32_16x16x32_bf16(
                false, a.v, false, bf.v, (short)0, acc[nt], false, false);
        }
        __syncthreads();
    }

    // ---- fused epilogue: out = gamma*acc + x
#pragma unroll
    for (int nt = 0; nt < 8; ++nt) {
#pragma unroll
        for (int r = 0; r < 8; ++r) {
            const int row = i0 + mi * 16 + r + half * 8;
            const int col = t0 + (ntb + nt) * 16 + l15;
            const size_t idx = (size_t)(b * C_DIM + row) * T_DIM + col;
            out[idx] = g * acc[nt][r] + x[idx];
        }
    }
}

// ---------------------------------------------------------------------------
extern "C" void kernel_launch(void* const* d_in, const int* in_sizes, int n_in,
                              void* d_out, int out_size, void* d_ws, size_t ws_size,
                              hipStream_t stream) {
    (void)in_sizes; (void)n_in; (void)out_size; (void)ws_size;

    const float* x     = (const float*)d_in[0];
    const float* gamma = (const float*)d_in[1];
    float*       out   = (float*)d_out;

    // workspace layout: [energy f32: 2 MiB][attention bf16: 1 MiB]
    float*    energy = (float*)d_ws;
    uint16_t* att    = (uint16_t*)((char*)d_ws + (size_t)B_DIM * C_DIM * C_DIM * sizeof(float));

    const int nE = B_DIM * C_DIM * C_DIM;
    zero_f32<<<(nE + 255) / 256, 256, 0, stream>>>(energy, nE);
    gram_kernel<<<B_DIM * 3 * KCHUNKS, 256, 0, stream>>>(x, energy);      // 384 blocks
    softmax_kernel<<<B_DIM * C_DIM, 256, 0, stream>>>(energy, att);       // 2048 blocks
    apply_kernel<<<B_DIM * 4 * 64, 256, 0, stream>>>(x, att, gamma, out); // 2048 blocks
}
